// NORM_net_62268435857445
// MI455X (gfx1250) — matrix operand order repaired and checked by reading, not compiled, not run on hardware
//
#include <hip/hip_runtime.h>
#include <hip/hip_bf16.h>
#include <math.h>

// ---------------- problem constants (from reference) ----------------
constexpr int Bn  = 4;
constexpr int Nn  = 8192;
constexpr int Cc  = 64;
constexpr int Mm  = 64;
constexpr int NK  = 6;
constexpr int Ll  = 4;
constexpr long RN = (long)Bn * Nn;        // 32768 rows of the token stream

typedef __attribute__((ext_vector_type(16))) __bf16 bf16x16;
typedef __attribute__((ext_vector_type(8)))  __bf16 bf16x8;
typedef __attribute__((ext_vector_type(8)))  float  f32x8;

// =====================================================================
// WMMA GEMM:  C[Mr,Nc] (f32) = A[Mr,K] (bf16, row-major) * B[K,Nc]
// where B is supplied PRE-TRANSPOSED: BT[Nc][K] (bf16, row-major).
// Tile 128x64, BK=32, 256 threads = 8 waves, each wave owns a 16x64 strip.
// Both tiles are staged with GLOBAL_LOAD_ASYNC_TO_LDS_B128 into a ping-pong
// LDS buffer (ASYNCcnt-tracked), software-pipelined one stage ahead.
// Fragment layouts follow CDNA5 ISA 7.12.2 (16-bit A 16x32, B 32x16, f32 C).
// =====================================================================
__global__ __launch_bounds__(256)
void k_gemm(const __bf16* __restrict__ A, const __bf16* __restrict__ BT,
            float* __restrict__ C, int Mr, int Nc, int K,
            int lda, int ldbt, int ldc)
{
    // padded row stride 40 (80B): 16B-aligned rows, conflict-free b128
    __shared__ __align__(16) __bf16 Al[2][128 * 40];
    __shared__ __align__(16) __bf16 Bl[2][64 * 40];

    const int tid  = threadIdx.x;
    const int lane = tid & 31;
    const int wave = tid >> 5;
    const int half = lane >> 4;     // 0: lanes 0-15, 1: lanes 16-31
    const int l16  = lane & 15;
    const int row0 = blockIdx.x * 128;
    const int col0 = blockIdx.y * 64;

    // per-thread staging coordinates (uniform, no divergence)
    const int ar0 = tid >> 2;              // A chunk 0 row (0..63)
    const int ac0 = tid & 3;               // A chunk col group
    const int ar1 = (tid + 256) >> 2;      // A chunk 1 row (64..127)
    const int bn  = tid >> 2;              // BT row (0..63)
    const int bc  = tid & 3;

    // issue one stage (3 async b128 per thread => 3 per wave):
    auto issue = [&](int buf, int k0) {
        int g0 = row0 + ar0; if (g0 >= Mr) g0 = Mr - 1;
        int g1 = row0 + ar1; if (g1 >= Mr) g1 = Mr - 1;
        unsigned d0 = (unsigned)(size_t)&Al[buf][ar0 * 40 + ac0 * 8];
        unsigned d1 = (unsigned)(size_t)&Al[buf][ar1 * 40 + ac0 * 8];
        unsigned d2 = (unsigned)(size_t)&Bl[buf][bn * 40 + bc * 8];
        unsigned long long s0 = (unsigned long long)(size_t)(A  + (size_t)g0 * lda + k0 + ac0 * 8);
        unsigned long long s1 = (unsigned long long)(size_t)(A  + (size_t)g1 * lda + k0 + ac0 * 8);
        unsigned long long s2 = (unsigned long long)(size_t)(BT + (size_t)(col0 + bn) * ldbt + k0 + bc * 8);
        asm volatile("global_load_async_to_lds_b128 %0, %1, off"
                     :: "v"(d0), "v"(s0) : "memory");
        asm volatile("global_load_async_to_lds_b128 %0, %1, off"
                     :: "v"(d1), "v"(s1) : "memory");
        asm volatile("global_load_async_to_lds_b128 %0, %1, off"
                     :: "v"(d2), "v"(s2) : "memory");
    };

    f32x8 zero = {0.f,0.f,0.f,0.f,0.f,0.f,0.f,0.f};
    f32x8 acc[4] = {zero, zero, zero, zero};

    const int nk = K >> 5;
    issue(0, 0);
    for (int i = 0; i < nk; ++i) {
        if (i + 1 < nk) {
            issue((i + 1) & 1, (i + 1) << 5);
            asm volatile("s_wait_asynccnt 3" ::: "memory");  // stage i landed
        } else {
            asm volatile("s_wait_asynccnt 0" ::: "memory");
        }
        __syncthreads();                                     // all waves landed

        const int buf = i & 1;
        // A fragment for this wave's 16-row strip (ISA 16-bit A 16x32 layout)
        const bf16x8* ar = (const bf16x8*)&Al[buf][(wave * 16 + l16) * 40];
        bf16x16 afr = __builtin_shufflevector(ar[half], ar[2 + half],
                       0,1,2,3,4,5,6,7,8,9,10,11,12,13,14,15);
        #pragma unroll
        for (int j = 0; j < 4; ++j) {
            // B fragment tile j: lanes 0-15 -> K 0..7,16..23, lanes 16-31 -> K 8..15,24..31
            const bf16x8* br = (const bf16x8*)&Bl[buf][(j * 16 + l16) * 40];
            bf16x16 bfr = __builtin_shufflevector(br[half], br[2 + half],
                           0,1,2,3,4,5,6,7,8,9,10,11,12,13,14,15);
            acc[j] = __builtin_amdgcn_wmma_f32_16x16x32_bf16(
                         false, afr, false, bfr, (short)0, acc[j], false, false);
        }
        __syncthreads();   // protect buf from stage i+2 overwrite
    }

    // store: f32 C/D layout: lane holds col l16, VGPR r -> row r + half*8
    #pragma unroll
    for (int j = 0; j < 4; ++j) {
        #pragma unroll
        for (int r = 0; r < 8; ++r) {
            int gr = row0 + wave * 16 + half * 8 + r;
            int gc = col0 + j * 16 + l16;
            if (gr < Mr) C[(size_t)gr * ldc + gc] = acc[j][r];
        }
    }
}

// ---------------- elementwise / small kernels ----------------
__global__ void k_cvt(const float* __restrict__ s, __bf16* __restrict__ d, long n) {
    long i  = (long)blockIdx.x * blockDim.x + threadIdx.x;
    long st = (long)gridDim.x * blockDim.x;
    for (; i < n; i += st) d[i] = (__bf16)s[i];
}

// transpose-convert: src f32 [K][Ncols] -> dst bf16 [Ncols][K]
__global__ void k_cvt_t(const float* __restrict__ s, __bf16* __restrict__ d,
                        int K, int Ncols) {
    long i  = (long)blockIdx.x * blockDim.x + threadIdx.x;
    long st = (long)gridDim.x * blockDim.x;
    long tot = (long)K * Ncols;
    for (; i < tot; i += st) {
        long k = i / Ncols; int c = (int)(i % Ncols);
        d[(size_t)c * K + k] = (__bf16)s[i];
    }
}

__device__ __forceinline__ float gelu_erf(float v) {
    return 0.5f * v * (1.0f + erff(v * 0.70710678118654752f));
}

// y = act(y + bias); bias may be null. mode: 0 none, 1 relu, 2 gelu
__global__ void k_bias(float* __restrict__ y, const float* __restrict__ bias,
                       long total, int cols, int mode) {
    long i  = (long)blockIdx.x * blockDim.x + threadIdx.x;
    long st = (long)gridDim.x * blockDim.x;
    for (; i < total; i += st) {
        float v = y[i];
        if (bias) v += bias[i % cols];
        if (mode == 1) v = v > 0.f ? v : 0.f;
        else if (mode == 2) v = gelu_erf(v);
        y[i] = v;
    }
}

// h += y + bias
__global__ void k_bias_res(float* __restrict__ h, const float* __restrict__ y,
                           const float* __restrict__ bias, long total, int cols) {
    long i  = (long)blockIdx.x * blockDim.x + threadIdx.x;
    long st = (long)gridDim.x * blockDim.x;
    for (; i < total; i += st) h[i] += y[i] + bias[i % cols];
}

// h += a + b
__global__ void k_add3(float* __restrict__ h, const float* __restrict__ a,
                       const float* __restrict__ b, long total) {
    long i  = (long)blockIdx.x * blockDim.x + threadIdx.x;
    long st = (long)gridDim.x * blockDim.x;
    for (; i < total; i += st) h[i] += a[i] + b[i];
}

// LayerNorm over 64 channels; one 64-thread block per row. In-place safe.
__global__ void k_ln(const float* __restrict__ x, float* __restrict__ y,
                     const float* __restrict__ g, const float* __restrict__ b) {
    __shared__ float sm[64];
    const long r = blockIdx.x;
    const int  t = threadIdx.x;
    float v = x[r * 64 + t];
    sm[t] = v; __syncthreads();
    for (int s = 32; s > 0; s >>= 1) { if (t < s) sm[t] += sm[t + s]; __syncthreads(); }
    float mean = sm[0] * (1.0f / 64.0f);
    __syncthreads();
    float d = v - mean;
    sm[t] = d * d; __syncthreads();
    for (int s = 32; s > 0; s >>= 1) { if (t < s) sm[t] += sm[t + s]; __syncthreads(); }
    float var = sm[0] * (1.0f / 64.0f);
    y[r * 64 + t] = d * rsqrtf(var + 1e-5f) * g[t] + b[t];
}

// stem: h0[(b*N+n)*64+c] = x[b,n]*w[0,c] + mesh[n,j]*w[1+j,c] + bias[c]
__global__ void k_stem(const float* __restrict__ x, const float* __restrict__ mesh,
                       const float* __restrict__ w, const float* __restrict__ bias,
                       float* __restrict__ out) {
    long i  = (long)blockIdx.x * blockDim.x + threadIdx.x;
    long st = (long)gridDim.x * blockDim.x;
    long tot = RN * 64;
    for (; i < tot; i += st) {
        long rn = i >> 6; int c = (int)(i & 63);
        int  n  = (int)(rn % Nn);
        float v = x[rn] * w[c]
                + mesh[n * 3 + 0] * w[64 + c]
                + mesh[n * 3 + 1] * w[128 + c]
                + mesh[n * 3 + 2] * w[192 + c]
                + bias[c];
        out[i] = v;
    }
}

// encoder first layer: e1[n,c] = gelu(mesh[n,:3] @ w[3,64] + b)
__global__ void k_enc_p1(const float* __restrict__ mesh, const float* __restrict__ w,
                         const float* __restrict__ bias, float* __restrict__ out) {
    long i  = (long)blockIdx.x * blockDim.x + threadIdx.x;
    long st = (long)gridDim.x * blockDim.x;
    long tot = (long)Nn * 64;
    for (; i < tot; i += st) {
        long n = i >> 6; int c = (int)(i & 63);
        float v = mesh[n * 3 + 0] * w[c]
                + mesh[n * 3 + 1] * w[64 + c]
                + mesh[n * 3 + 2] * w[128 + c]
                + bias[c];
        out[i] = gelu_erf(v);
    }
}

// eo[n,j] = a[n,:128] @ w[:128,j] + b[j]   (Nc=6, too skinny for WMMA)
__global__ void k_o2(const float* __restrict__ a, const float* __restrict__ w,
                     const float* __restrict__ bias, float* __restrict__ out) {
    long i  = (long)blockIdx.x * blockDim.x + threadIdx.x;
    long st = (long)gridDim.x * blockDim.x;
    long tot = (long)Nn * NK;
    for (; i < tot; i += st) {
        long n = i / NK; int j = (int)(i % NK);
        float acc = bias[j];
        const float* ar = a + n * 128;
        for (int t = 0; t < 128; ++t) acc += ar[t] * w[t * NK + j];
        out[i] = acc;
    }
}

// glog[m,j] = sum_n inv[m,n] * eo[n,j]   (384 block-reductions over K=8192)
__global__ void k_inv_e(const float* __restrict__ inv, const float* __restrict__ eo,
                        float* __restrict__ out) {
    __shared__ float red[256];
    int m = blockIdx.x / NK, j = blockIdx.x % NK;
    const float* ir = inv + (size_t)m * Nn;
    float acc = 0.f;
    for (int n = threadIdx.x; n < Nn; n += 256) acc += ir[n] * eo[n * NK + j];
    red[threadIdx.x] = acc; __syncthreads();
    for (int s = 128; s > 0; s >>= 1) {
        if (threadIdx.x < s) red[threadIdx.x] += red[threadIdx.x + s];
        __syncthreads();
    }
    if (threadIdx.x == 0) out[m * NK + j] = red[0];
}

// row softmax of [64, 6] in place (one thread per row)
__global__ void k_softmax(float* __restrict__ a) {
    int m = threadIdx.x;
    if (m >= Mm) return;
    float* r = a + m * NK;
    float mx = r[0];
    for (int j = 1; j < NK; ++j) mx = fmaxf(mx, r[j]);
    float s = 0.f;
    for (int j = 0; j < NK; ++j) { float e = expf(r[j] - mx); r[j] = e; s += e; }
    float inv = 1.0f / s;
    for (int j = 0; j < NK; ++j) r[j] *= inv;
}

// w[i,o,m] = sum_k spec_w[k,i,o,m] * geo[m,k]
__global__ void k_spec_w(const float* __restrict__ sw, const float* __restrict__ geo,
                         float* __restrict__ w) {
    long i  = (long)blockIdx.x * blockDim.x + threadIdx.x;
    long st = (long)gridDim.x * blockDim.x;
    long tot = (long)Cc * Cc * Mm;
    for (; i < tot; i += st) {
        int m  = (int)(i & 63);
        int o  = (int)((i >> 6) & 63);
        int ii = (int)(i >> 12);
        float acc = 0.f;
        #pragma unroll
        for (int k = 0; k < NK; ++k)
            acc += sw[(((size_t)k * Cc + ii) * Cc + o) * Mm + m] * geo[m * NK + k];
        w[i] = acc;
    }
}

// Q[b,m,i] = sum_n inv[m,n] * z[b,n,i]   (block per (b,m); 64 i-lanes x 4 K-chunks)
__global__ void k_cm(const float* __restrict__ inv, const float* __restrict__ z,
                     float* __restrict__ Q) {
    __shared__ float red[256];
    int bm = blockIdx.x;            // 0..255
    int b  = bm >> 6, m = bm & 63;
    int ii = threadIdx.x & 63, ch = threadIdx.x >> 6;
    const float* ir = inv + (size_t)m * Nn;
    const float* zb = z + (size_t)b * Nn * 64;
    float acc = 0.f;
    int n0 = ch * (Nn / 4), n1 = n0 + (Nn / 4);
    for (int n = n0; n < n1; ++n) acc += ir[n] * zb[(size_t)n * 64 + ii];
    red[threadIdx.x] = acc; __syncthreads();
    if (ch == 0)
        Q[((size_t)b * 64 + m) * 64 + ii] =
            red[ii] + red[64 + ii] + red[128 + ii] + red[192 + ii];
}

// R[b,m,o] = sum_i Q[b,m,i] * w[i,o,m]
__global__ void k_cg(const float* __restrict__ Q, const float* __restrict__ w,
                     float* __restrict__ R) {
    long i  = (long)blockIdx.x * blockDim.x + threadIdx.x;
    long st = (long)gridDim.x * blockDim.x;
    long tot = (long)Bn * 64 * 64;
    for (; i < tot; i += st) {
        int o = (int)(i & 63);
        int m = (int)((i >> 6) & 63);
        const float* q = Q + (i >> 6 << 6);   // (b*64+m)*64
        float acc = 0.f;
        for (int ii = 0; ii < 64; ++ii) acc += q[ii] * w[((size_t)ii * 64 + o) * 64 + m];
        R[i] = acc;
    }
}

// out[r] = t[r,:128] @ w2 + b2
__global__ void k_fc2(const float* __restrict__ t, const float* __restrict__ w2,
                      const float* __restrict__ b2, float* __restrict__ out) {
    long i  = (long)blockIdx.x * blockDim.x + threadIdx.x;
    long st = (long)gridDim.x * blockDim.x;
    for (; i < RN; i += st) {
        const float* tr = t + i * 128;
        float acc = b2[0];
        for (int j = 0; j < 128; ++j) acc += tr[j] * w2[j];
        out[i] = acc;
    }
}

// =====================================================================
extern "C" void kernel_launch(void* const* d_in, const int* in_sizes, int n_in,
                              void* d_out, int out_size, void* d_ws, size_t ws_size,
                              hipStream_t stream) {
    (void)in_sizes; (void)n_in; (void)out_size; (void)ws_size;

    const float* x        = (const float*)d_in[0];
    const float* MAT      = (const float*)d_in[1];
    const float* INV      = (const float*)d_in[2];
    const float* mesh     = (const float*)d_in[3];
    const float* adj      = (const float*)d_in[4];
    const float* fc0_w    = (const float*)d_in[5];
    const float* fc0_b    = (const float*)d_in[6];
    const float* fc01_w   = (const float*)d_in[7];
    const float* fc01_b   = (const float*)d_in[8];
    const float* bln1_g   = (const float*)d_in[9];
    const float* bln1_b   = (const float*)d_in[10];
    const float* n1_g     = (const float*)d_in[11];
    const float* n1_b     = (const float*)d_in[12];
    const float* spec_w   = (const float*)d_in[13];
    const float* ep1_w    = (const float*)d_in[14];
    const float* ep1_b    = (const float*)d_in[15];
    const float* ep2_w    = (const float*)d_in[16];
    const float* ep2_b    = (const float*)d_in[17];
    const float* eg1_w    = (const float*)d_in[18];
    const float* eg1_b    = (const float*)d_in[19];
    const float* eg2_w    = (const float*)d_in[20];
    const float* eg2_b    = (const float*)d_in[21];
    const float* eo1_w    = (const float*)d_in[22];
    const float* eo1_b    = (const float*)d_in[23];
    const float* eo2_w    = (const float*)d_in[24];
    const float* eo2_b    = (const float*)d_in[25];
    const float* sg1_w    = (const float*)d_in[26];
    const float* sg1_b    = (const float*)d_in[27];
    const float* sg2_w    = (const float*)d_in[28];
    const float* sg2_b    = (const float*)d_in[29];
    const float* bln2_g   = (const float*)d_in[30];
    const float* bln2_b   = (const float*)d_in[31];
    const float* mpre_w   = (const float*)d_in[32];
    const float* mpre_b   = (const float*)d_in[33];
    const float* mpost_w  = (const float*)d_in[34];
    const float* mpost_b  = (const float*)d_in[35];
    const float* ln1_g    = (const float*)d_in[36];
    const float* ln1_b    = (const float*)d_in[37];
    const float* fc1_w    = (const float*)d_in[38];
    const float* fc1_b    = (const float*)d_in[39];
    const float* fc2_w    = (const float*)d_in[40];
    const float* fc2_b    = (const float*)d_in[41];
    float* out = (float*)d_out;

    // --------- workspace bump allocator ---------
    char*  wsb = (char*)d_ws;
    size_t off = 0;
    auto alloc = [&](size_t bytes) -> void* {
        off = (off + 255) & ~(size_t)255;
        void* p = wsb + off;
        off += bytes;
        return p;
    };

    __bf16* adjb = (__bf16*)alloc((size_t)Nn * Nn * 2);      // 134 MB -> fits L2
    __bf16* mpb  = (__bf16*)alloc((size_t)Nn * Mm * 2);
    float*  h    = (float*)alloc(RN * 64 * 4);
    float*  z    = (float*)alloc(RN * 64 * 4);
    __bf16* zb   = (__bf16*)alloc(RN * 128 * 2);   // plain bf16 (A operand)
    __bf16* tbp  = (__bf16*)alloc(RN * 128 * 2);   // plain bf16 (A operand)
    __bf16* tbT  = (__bf16*)alloc(RN * 128 * 2);   // transposed bf16 (B operand)
    float*  t    = (float*)alloc(RN * 128 * 4);
    float*  u    = (float*)alloc(RN * 128 * 4);
    float*  x1   = (float*)alloc(RN * 64 * 4);
    float*  x2   = (float*)alloc(RN * 64 * 4);
    float*  e    = (float*)alloc((size_t)Nn * 64 * 4);
    float*  ea   = (float*)alloc((size_t)Nn * 128 * 4);
    __bf16* eb16 = (__bf16*)alloc((size_t)Nn * 128 * 2);  // plain (A operand)
    __bf16* ebT  = (__bf16*)alloc((size_t)Nn * 128 * 2);  // transposed (B operand)
    float*  eo   = (float*)alloc((size_t)Nn * NK * 4);
    float*  glog = (float*)alloc(Mm * NK * 4);
    float*  Q    = (float*)alloc((size_t)Bn * 64 * 64 * 4);
    float*  wsp  = (float*)alloc((size_t)Cc * Cc * Mm * 4);
    float*  R    = (float*)alloc((size_t)Bn * 64 * 64 * 4);
    __bf16* Rb   = (__bf16*)alloc((size_t)Bn * 64 * 64 * 2); // transposed per batch
    __bf16* wb   = (__bf16*)alloc(65536);                    // weight staging (T)

    // --------- launch helpers ---------
    auto gemm = [&](const __bf16* A, const __bf16* BTp, float* Cp,
                    int Mr, int Nc, int K, int lda, int ldbt, int ldc) {
        dim3 g((Mr + 127) / 128, Nc / 64);
        k_gemm<<<g, 256, 0, stream>>>(A, BTp, Cp, Mr, Nc, K, lda, ldbt, ldc);
    };
    auto cvt = [&](const float* s, __bf16* d, long n) {
        long bl = (n + 255) / 256; if (bl > 8192) bl = 8192;
        k_cvt<<<(unsigned)bl, 256, 0, stream>>>(s, d, n);
    };
    auto cvt_t = [&](const float* s, __bf16* d, int K, int Ncols) {
        long n = (long)K * Ncols;
        long bl = (n + 255) / 256; if (bl > 8192) bl = 8192;
        k_cvt_t<<<(unsigned)bl, 256, 0, stream>>>(s, d, K, Ncols);
    };
    auto ew = [&](long n) {
        long bl = (n + 255) / 256; if (bl > 16384) bl = 16384;
        return dim3((unsigned)bl);
    };
    auto bias = [&](float* y, const float* bp, long rows, int cols, int mode) {
        k_bias<<<ew(rows * cols), 256, 0, stream>>>(y, bp, rows * cols, cols, mode);
    };

    // --------- one-time: adj -> bf16 (L2-resident), MATRIX_Phy -> bf16
    cvt(adj, adjb, (long)Nn * Nn);
    cvt(MAT, mpb, (long)Nn * Mm);

    // --------- stem: h = (concat(x,mesh) @ fc0 + b0) @ fc01 + b01
    k_stem<<<ew(RN * 64), 256, 0, stream>>>(x, mesh, fc0_w, fc0_b, z);
    cvt(z, zb, RN * 64);
    cvt_t(fc01_w, wb, 64, 64);
    gemm(zb, wb, h, (int)RN, 64, 64, 64, 64, 64);
    bias(h, fc01_b, RN, 64, 0);

    for (int l = 0; l < Ll; ++l) {
        // z = LN1(h)
        k_ln<<<(unsigned)RN, 64, 0, stream>>>(h, z, bln1_g + l * 64, bln1_b + l * 64);

        // ---------- MeshGCN encoder -> geoT [M,NK] ----------
        k_enc_p1<<<ew((long)Nn * 64), 256, 0, stream>>>(mesh, ep1_w + l * 192, ep1_b + l * 64, ea);
        cvt(ea, eb16, (long)Nn * 64);
        cvt_t(ep2_w + l * 4096, wb, 64, 64);
        gemm(eb16, wb, e, Nn, 64, 64, 64, 64, 64);
        bias(e, ep2_b + l * 64, Nn, 64, 0);

        cvt(e, eb16, (long)Nn * 64);
        cvt_t(eg1_w + l * 4096, wb, 64, 64);
        gemm(eb16, wb, ea, Nn, 64, 64, 64, 64, 64);
        bias(ea, eg1_b + l * 64, Nn, 64, 0);
        cvt_t(ea, ebT, Nn, 64);                          // B operand for adj
        gemm(adjb, ebT, e, Nn, 64, Nn, Nn, Nn, 64);      // adj @ (...)
        bias(e, nullptr, Nn, 64, 2);                     // gelu

        cvt(e, eb16, (long)Nn * 64);
        cvt_t(eg2_w + l * 4096, wb, 64, 64);
        gemm(eb16, wb, ea, Nn, 64, 64, 64, 64, 64);
        bias(ea, eg2_b + l * 64, Nn, 64, 0);
        cvt_t(ea, ebT, Nn, 64);
        gemm(adjb, ebT, e, Nn, 64, Nn, Nn, Nn, 64);      // adj @ (...)

        cvt(e, eb16, (long)Nn * 64);
        cvt_t(eo1_w + l * 8192, wb, 64, 128);
        gemm(eb16, wb, ea, Nn, 128, 64, 64, 64, 128);
        bias(ea, eo1_b + l * 128, Nn, 128, 0);
        k_o2<<<ew((long)Nn * NK), 256, 0, stream>>>(ea, eo2_w + l * 768, eo2_b + l * NK, eo);
        k_inv_e<<<Mm * NK, 256, 0, stream>>>(INV, eo, glog);
        k_softmax<<<1, 64, 0, stream>>>(glog);

        // ---------- geodesic-modulated spectral conv ----------
        k_cm<<<Bn * 64, 256, 0, stream>>>(INV, z, Q);                         // Q[b,m,i]
        k_spec_w<<<ew((long)Cc * Cc * Mm), 256, 0, stream>>>(
            spec_w + (size_t)l * NK * Cc * Cc * Mm, glog, wsp);               // w[i,o,m]
        k_cg<<<ew((long)Bn * 64 * 64), 256, 0, stream>>>(Q, wsp, R);          // R[b,m,o]
        for (int b = 0; b < Bn; ++b)
            cvt_t(R + (size_t)b * 4096, Rb + (size_t)b * 4096, 64, 64);
        for (int b = 0; b < Bn; ++b)
            gemm(mpb, Rb + (size_t)b * 4096, x1 + (size_t)b * Nn * 64,
                 Nn, 64, 64, 64, 64, 64);                                     // x1_b = MAT @ R_b
        k_ln<<<(unsigned)RN, 64, 0, stream>>>(x1, x1, n1_g + l * 64, n1_b + l * 64);

        // ---------- spatial GCN branch ----------
        cvt(z, zb, RN * 64);
        cvt_t(sg1_w + l * 8192, wb, 64, 128);
        gemm(zb, wb, t, (int)RN, 128, 64, 64, 64, 128);
        bias(t, sg1_b + l * 128, RN, 128, 0);
        for (int b = 0; b < Bn; ++b)
            cvt_t(t + (size_t)b * Nn * 128, tbT + (size_t)b * Nn * 128, Nn, 128);
        for (int b = 0; b < Bn; ++b)
            gemm(adjb, tbT + (size_t)b * Nn * 128, u + (size_t)b * Nn * 128,
                 Nn, 128, Nn, Nn, Nn, 128);
        bias(u, nullptr, RN, 128, 1);                    // relu
        cvt(u, tbp, RN * 128);
        cvt_t(sg2_w + l * 8192, wb, 128, 64);
        gemm(tbp, wb, t, (int)RN, 64, 128, 128, 128, 64);
        bias(t, sg2_b + l * 64, RN, 64, 0);
        for (int b = 0; b < Bn; ++b)
            cvt_t(t + (size_t)b * Nn * 64, tbT + (size_t)b * Nn * 64, Nn, 64);
        for (int b = 0; b < Bn; ++b)
            gemm(adjb, tbT + (size_t)b * Nn * 64, x2 + (size_t)b * Nn * 64,
                 Nn, 64, Nn, Nn, Nn, 64);

        k_add3<<<ew(RN * 64), 256, 0, stream>>>(h, x1, x2, RN * 64);

        // ---------- MLP ----------
        k_ln<<<(unsigned)RN, 64, 0, stream>>>(h, z, bln2_g + l * 64, bln2_b + l * 64);
        cvt(z, zb, RN * 64);
        cvt_t(mpre_w + l * 8192, wb, 64, 128);
        gemm(zb, wb, t, (int)RN, 128, 64, 64, 64, 128);
        bias(t, mpre_b + l * 128, RN, 128, 2);           // gelu
        cvt(t, tbp, RN * 128);
        cvt_t(mpost_w + l * 8192, wb, 128, 64);
        gemm(tbp, wb, u, (int)RN, 64, 128, 128, 128, 64);
        k_bias_res<<<ew(RN * 64), 256, 0, stream>>>(h, u, mpost_b + l * 64, RN * 64, 64);
    }

    // --------- head ---------
    k_ln<<<(unsigned)RN, 64, 0, stream>>>(h, z, ln1_g, ln1_b);
    cvt(z, zb, RN * 64);
    cvt_t(fc1_w, wb, 64, 128);
    gemm(zb, wb, t, (int)RN, 128, 64, 64, 64, 128);
    bias(t, fc1_b, RN, 128, 2);                          // gelu
    k_fc2<<<ew(RN), 256, 0, stream>>>(t, fc2_w, fc2_b, out);
}
